// GptOssAttention_75041668596192
// MI455X (gfx1250) — matrix operand order repaired and checked by reading, not compiled, not run on hardware
//
#include <hip/hip_runtime.h>

// ---------------------------------------------------------------------------
// GQA attention block for MI455X (gfx1250, wave32, WMMA + TDM).
// B=2, S=2048, D=4096, H=32, KVH=8, HD=128, groups=4.
// GEMM stages: v_wmma_f32_16x16x32_bf16 (bf16 in, f32 accum).
// Tile movement: tensor_load_to_lds (TDM) where the source is already bf16,
// register-pipelined global loads + hardware f32->bf16 convert elsewhere.
// ---------------------------------------------------------------------------

#define CB   2
#define CS   2048
#define CD   4096
#define CH   32
#define CKVH 8
#define CHD  128
#define CM   (CB * CS)            // 4096 rows of (b,s)
#define CQN  (CH * CHD)           // 4096
#define CKVN (CKVH * CHD)         // 1024
#define NEGV (-1.0e9f)

typedef __attribute__((ext_vector_type(16))) __bf16 bf16x16;
typedef __attribute__((ext_vector_type(8)))  float  f32x8;
typedef __attribute__((ext_vector_type(4)))  unsigned int u32x4;
typedef __attribute__((ext_vector_type(8)))  int    i32x8;
typedef __attribute__((ext_vector_type(4)))  int    i32x4;

#if defined(__gfx1250__) && __has_builtin(__builtin_amdgcn_tensor_load_to_lds) && \
    __has_builtin(__builtin_amdgcn_s_wait_tensorcnt)
#define HAVE_TDM 1
#else
#define HAVE_TDM 0
#endif

// Hardware f32->bf16 (RNE on gfx1250; compiler emits v_cvt_*_bf16_f32).
__device__ __forceinline__ unsigned short f2bf(float f) {
    union { __bf16 b; unsigned short u; } cv;
    cv.b = (__bf16)f;
    return cv.u;
}
// Packed pair -> one 32-bit value (lets clang fuse into v_cvt_pk_bf16_f32).
__device__ __forceinline__ unsigned f2bf2(float lo, float hi) {
    union { __bf16 b[2]; unsigned u; } cv;
    cv.b[0] = (__bf16)lo;
    cv.b[1] = (__bf16)hi;
    return cv.u;
}
__device__ __forceinline__ float bf2f(unsigned short h) {
    return __uint_as_float(((unsigned)h) << 16);
}
__device__ __forceinline__ f32x8 zero8() {
    f32x8 z;
#pragma unroll
    for (int i = 0; i < 8; ++i) z[i] = 0.0f;
    return z;
}
__device__ __forceinline__ f32x8 wmma_bf16(bf16x16 a, bf16x16 b, f32x8 c) {
    return __builtin_amdgcn_wmma_f32_16x16x32_bf16(
        false, a, false, b, (short)0, c, false, false);
}
__device__ __forceinline__ unsigned lds_off(const void* p) {
    // LDS aperture keeps the wave-relative byte offset in addr[31:0].
    return (unsigned)(size_t)p;
}

#if HAVE_TDM
// 2-D bf16 tile DMA: global (row-major, stride0 elems/row) -> LDS (packed).
// D# per CDNA5 ISA 8.3/8.4: group0 = {count=1, lds_addr, global_addr, type=2},
// group1 = {data_size=2B, tensor dims, tile dims, dim0 stride}, groups2/3 = 0.
// This toolchain exposes the 6-arg builtin: (g0, g1, g2, g3, g_extra, cpol).
__device__ __forceinline__ void tdm_load_2d(unsigned lds_byte_off, const void* g,
                                            unsigned tile0, unsigned tile1,
                                            unsigned t0, unsigned t1,
                                            unsigned stride0) {
    const unsigned long long ga = (unsigned long long)(size_t)g;
    u32x4 g0 = {1u,                                   // count=1 (user mode)
                lds_byte_off,
                (unsigned)ga,
                ((unsigned)((ga >> 32) & 0x01FFFFFFull)) | 0x80000000u}; // type=2
    i32x8 g1;
    g1[0] = 0x00010000;                               // data_size = 1 -> 2 bytes
    g1[1] = (int)((t0 & 0xFFFFu) << 16);              // tensor_dim0 lo
    g1[2] = (int)((t0 >> 16) | ((t1 & 0xFFFFu) << 16));
    g1[3] = (int)((t1 >> 16) | (tile0 << 16));        // tile_dim0
    g1[4] = (int)(tile1 & 0xFFFFu);                   // tile_dim1 (tile_dim2=0)
    g1[5] = (int)stride0;                             // tensor_dim0_stride lo
    g1[6] = 0;
    g1[7] = 0;
    const i32x4 gz4 = {0, 0, 0, 0};
    const i32x8 gz8 = {0, 0, 0, 0, 0, 0, 0, 0};
    __builtin_amdgcn_tensor_load_to_lds(g0, g1, gz4, gz4, gz8, 0);
}
#endif

// A fragment (16x32 bf16) from row-major LDS tile [row][ld], base (r0, k0).
__device__ __forceinline__ bf16x16 load_a_frag(const unsigned short* lds,
                                               int r0, int k0, int ld, int lane) {
    const int row  = r0 + (lane & 15);
    const int koff = ((lane >> 4) & 1) * 16;
    union { unsigned u[8]; bf16x16 v; } t;
    const unsigned short* p = lds + row * ld + k0;
#pragma unroll
    for (int i = 0; i < 8; ++i) {
        const int kk = koff + 2 * i;
        t.u[i] = *(const unsigned*)(p + kk);
    }
    return t.v;
}

// B fragment (32x16 bf16) from LDS stored transposed [n][k] (k contiguous).
__device__ __forceinline__ bf16x16 load_b_frag(const unsigned short* lds,
                                               int n0, int k0, int ld, int lane) {
    const int col = n0 + (lane & 15);
    const int off = ((lane >> 4) & 1) * 16;
    union { unsigned u[8]; bf16x16 v; } t;
    const unsigned short* p = lds + col * ld + k0 + off;
#pragma unroll
    for (int i = 0; i < 8; ++i) t.u[i] = *(const unsigned*)(p + 2 * i);
    return t.v;
}

// ---------------------------------------------------------------------------
// Kernel 1: fused QKV projection, software-pipelined register staging.
// ---------------------------------------------------------------------------
__global__ __launch_bounds__(256)
void qkv_gemm(const float* __restrict__ X,
              const float* __restrict__ Wq,
              const float* __restrict__ Wk,
              const float* __restrict__ Wv,
              unsigned short* __restrict__ qws,
              unsigned short* __restrict__ kws,
              unsigned short* __restrict__ vws) {
    __shared__ unsigned short As[128 * 32];  // [m][k]
    __shared__ unsigned short Bs[128 * 32];  // [n][k] transposed

    const int t    = threadIdx.x;
    const int lane = t & 31;
    const int wid  = t >> 5;
    const int wr   = wid >> 1;
    const int wc   = wid & 1;
    const int m0   = blockIdx.y * 128;
    const int n0   = blockIdx.x * 128;

    const float* Wp;  int ldw, nn, region;
    if (n0 < CQN)             { Wp = Wq; ldw = CQN;  nn = n0;               region = 0; }
    else if (n0 < CQN + CKVN) { Wp = Wk; ldw = CKVN; nn = n0 - CQN;         region = 1; }
    else                      { Wp = Wv; ldw = CKVN; nn = n0 - CQN - CKVN;  region = 2; }

    float4 fa[4], fb[4];
    auto loadA = [&](int kk) {
#pragma unroll
        for (int i = 0; i < 4; ++i) {
            const int idx = t + i * 256;
            const int row = idx >> 3, c4 = idx & 7;
            fa[i] = *(const float4*)(X + (size_t)(m0 + row) * CD + kk + c4 * 4);
        }
    };
    auto loadB = [&](int kk) {
#pragma unroll
        for (int i = 0; i < 4; ++i) {
            const int idx = t + i * 256;
            const int row = idx >> 5, c4 = idx & 31;
            fb[i] = *(const float4*)(Wp + (size_t)(kk + row) * ldw + nn + c4 * 4);
        }
    };
    auto storeA = [&]() {
#pragma unroll
        for (int i = 0; i < 4; ++i) {
            const int idx = t + i * 256;
            const int row = idx >> 3, c4 = idx & 7;
            unsigned* d = (unsigned*)(As + row * 32 + c4 * 4);
            d[0] = f2bf2(fa[i].x, fa[i].y);
            d[1] = f2bf2(fa[i].z, fa[i].w);
        }
    };
    auto storeB = [&]() {
#pragma unroll
        for (int i = 0; i < 4; ++i) {
            const int idx = t + i * 256;
            const int row = idx >> 5, c4 = idx & 31;
            const int cb = c4 * 4;
            Bs[(cb + 0) * 32 + row] = f2bf(fb[i].x);
            Bs[(cb + 1) * 32 + row] = f2bf(fb[i].y);
            Bs[(cb + 2) * 32 + row] = f2bf(fb[i].z);
            Bs[(cb + 3) * 32 + row] = f2bf(fb[i].w);
        }
    };

    f32x8 c[2][4];
#pragma unroll
    for (int i = 0; i < 2; ++i)
#pragma unroll
        for (int j = 0; j < 4; ++j) c[i][j] = zero8();

    loadA(0); loadB(0);
    for (int k0 = 0; k0 < CD; k0 += 32) {
        storeA(); storeB();
        __syncthreads();
        if (k0 + 32 < CD) { loadA(k0 + 32); loadB(k0 + 32); }  // overlap w/ WMMA

        bf16x16 a[2], b[4];
#pragma unroll
        for (int si = 0; si < 2; ++si)
            a[si] = load_a_frag(As, wr * 32 + si * 16, 0, 32, lane);
#pragma unroll
        for (int sj = 0; sj < 4; ++sj)
            b[sj] = load_b_frag(Bs, wc * 64 + sj * 16, 0, 32, lane);
#pragma unroll
        for (int si = 0; si < 2; ++si)
#pragma unroll
            for (int sj = 0; sj < 4; ++sj)
                c[si][sj] = wmma_bf16(a[si], b[sj], c[si][sj]);
        __syncthreads();
    }

    const int half8 = ((lane >> 4) & 1) * 8;
#pragma unroll
    for (int si = 0; si < 2; ++si)
#pragma unroll
        for (int sj = 0; sj < 4; ++sj)
#pragma unroll
            for (int r = 0; r < 8; ++r) {
                const int gm = m0 + wr * 32 + si * 16 + r + half8;
                const int gn = n0 + wc * 64 + sj * 16 + (lane & 15);
                const int bb = gm >> 11, ss = gm & 2047;
                const unsigned short val = f2bf(c[si][sj][r]);
                if (region == 0) {
                    const int h = gn >> 7, dd = gn & 127;
                    qws[(((size_t)(bb * CH + h)) * CS + ss) * CHD + dd] = val;
                } else if (region == 1) {
                    const int g2 = gn - CQN, h = g2 >> 7, dd = g2 & 127;
                    kws[(((size_t)(bb * CKVH + h)) * CS + ss) * CHD + dd] = val;
                } else {
                    const int g2 = gn - CQN - CKVN, h = g2 >> 7, dd = g2 & 127;
                    vws[(((size_t)(bb * CKVH + h)) * CS + ss) * CHD + dd] = val;
                }
            }
}

// ---------------------------------------------------------------------------
// Kernel 2: RoPE in place on bf16 q/k workspaces.
// ---------------------------------------------------------------------------
__global__ __launch_bounds__(256)
void rope_kernel(unsigned short* __restrict__ qws,
                 unsigned short* __restrict__ kws,
                 const float* __restrict__ sins,
                 const float* __restrict__ coss) {
    const int QTOT = CB * CH * CS * (CHD / 2);
    const int id = blockIdx.x * 256 + threadIdx.x;
    unsigned short* p;
    int rem;
    if (id < QTOT) { p = qws; rem = id; }
    else           { p = kws; rem = id - QTOT; }
    const int j  = rem & 63;
    const int s  = (rem >> 6) & 2047;
    const int bh = rem >> 17;
    const size_t base = (((size_t)bh) * CS + s) * CHD;
    const float xr = bf2f(p[base + j]);
    const float xi = bf2f(p[base + 64 + j]);
    const float sn = sins[s * 64 + j];
    const float cs = coss[s * 64 + j];
    p[base + j]      = f2bf(xr * cs - xi * sn);
    p[base + 64 + j] = f2bf(xr * sn + xi * cs);
}

// ---------------------------------------------------------------------------
// Kernel 3: causal flash attention. Q/K tiles via TDM; V transposed on VALU.
// Dynamic LDS: Q(32K)+K(32K)+V^T(32K)+P(32K) = 128 KB (WGP has 320 KB).
// ---------------------------------------------------------------------------
__global__ __launch_bounds__(256)
void attn_kernel(const unsigned short* __restrict__ qws,
                 const unsigned short* __restrict__ kws,
                 const unsigned short* __restrict__ vws,
                 unsigned short* __restrict__ aws) {
    extern __shared__ unsigned short sm[];
    unsigned short* Qs = sm;               // [qrow][hd]
    unsigned short* Ks = sm + 16384;       // [key][hd]  (B-layout [n][k], k=hd)
    unsigned short* Vs = sm + 32768;       // [hd][key]  (B-layout [n][k], k=key)
    unsigned short* Ps = sm + 49152;       // [qrow][key] (A-layout)

    const int t    = threadIdx.x;
    const int lane = t & 31;
    const int wid  = t >> 5;
    const int qt   = blockIdx.x;
    const int h    = blockIdx.y;
    const int bb   = blockIdx.z;
    const int kvh  = h >> 2;
    const int qr0  = wid * 16;
    const int half8 = ((lane >> 4) & 1) * 8;
    const float scale = 0.08838834764831845f;

    const unsigned short* qsrc =
        qws + (((size_t)(bb * CH + h)) * CS + (size_t)qt * 128) * CHD;
#if HAVE_TDM
    if (wid == 0)
        tdm_load_2d(lds_off(Qs), qsrc, 128, 128, 128, 128, 128);
#else
#pragma unroll
    for (int i = 0; i < 8; ++i) {
        const int idx = t + i * 256;
        ((uint4*)Qs)[idx] = ((const uint4*)qsrc)[idx];
    }
#endif

    float mrow[8], lrow[8];
    f32x8 oacc[8];
#pragma unroll
    for (int r = 0; r < 8; ++r) { mrow[r] = -3.0e38f; lrow[r] = 0.0f; }
#pragma unroll
    for (int oj = 0; oj < 8; ++oj) oacc[oj] = zero8();

    for (int kt = 0; kt <= qt; ++kt) {
        __syncthreads();   // previous tile fully consumed
        const unsigned short* ksrc =
            kws + (((size_t)(bb * CKVH + kvh)) * CS + (size_t)kt * 128) * CHD;
        const unsigned short* vsrc =
            vws + (((size_t)(bb * CKVH + kvh)) * CS + (size_t)kt * 128) * CHD;
#if HAVE_TDM
        if (wid == 0)
            tdm_load_2d(lds_off(Ks), ksrc, 128, 128, 128, 128, 128);
#else
#pragma unroll
        for (int i = 0; i < 8; ++i) {
            const int idx = t + i * 256;
            ((uint4*)Ks)[idx] = ((const uint4*)ksrc)[idx];
        }
#endif
        // V tile: batch loads into regs, then transpose-scatter into LDS.
        uint4 vreg[8];
#pragma unroll
        for (int i = 0; i < 8; ++i) {
            const int idx = t + i * 256;
            const int row = idx >> 4, c8 = idx & 15;
            vreg[i] = ((const uint4*)(vsrc + (size_t)row * CHD))[c8];
        }
#pragma unroll
        for (int i = 0; i < 8; ++i) {
            const int idx = t + i * 256;
            const int row = idx >> 4, c8 = idx & 15;
            union { uint4 q; unsigned short e[8]; } u;
            u.q = vreg[i];
#pragma unroll
            for (int j = 0; j < 8; ++j) Vs[(c8 * 8 + j) * 128 + row] = u.e[j];
        }
#if HAVE_TDM
        if (wid == 0) __builtin_amdgcn_s_wait_tensorcnt(0);
#endif
        __syncthreads();

        // S = Q * K^T
        bf16x16 aq[4];
#pragma unroll
        for (int ks = 0; ks < 4; ++ks)
            aq[ks] = load_a_frag(Qs, qr0, ks * 32, 128, lane);
        f32x8 sacc[8];
#pragma unroll
        for (int sj = 0; sj < 8; ++sj) {
            sacc[sj] = zero8();
#pragma unroll
            for (int ks = 0; ks < 4; ++ks) {
                bf16x16 bk = load_b_frag(Ks, sj * 16, ks * 32, 128, lane);
                sacc[sj] = wmma_bf16(aq[ks], bk, sacc[sj]);
            }
        }

        // scale + causal mask
        const int qrow_base = qt * 128 + qr0 + half8;
        const int key_col   = kt * 128 + (lane & 15);
#pragma unroll
        for (int sj = 0; sj < 8; ++sj)
#pragma unroll
            for (int r = 0; r < 8; ++r) {
                float s = sacc[sj][r] * scale;
                if (key_col + sj * 16 > qrow_base + r) s = NEGV;
                sacc[sj][r] = s;
            }

        // online softmax (row stats live in 16-lane halves of the wave32)
        float mnew[8], alpha[8], rs[8];
#pragma unroll
        for (int r = 0; r < 8; ++r) {
            float mx = sacc[0][r];
#pragma unroll
            for (int sj = 1; sj < 8; ++sj) mx = fmaxf(mx, sacc[sj][r]);
#pragma unroll
            for (int off = 1; off < 16; off <<= 1)
                mx = fmaxf(mx, __shfl_xor(mx, off, 32));
            mnew[r]  = fmaxf(mrow[r], mx);
            alpha[r] = __expf(mrow[r] - mnew[r]);
            rs[r]    = 0.0f;
        }
#pragma unroll
        for (int sj = 0; sj < 8; ++sj)
#pragma unroll
            for (int r = 0; r < 8; ++r) {
                const float pv = __expf(sacc[sj][r] - mnew[r]);
                sacc[sj][r] = pv;
                rs[r] += pv;
            }
#pragma unroll
        for (int r = 0; r < 8; ++r) {
#pragma unroll
            for (int off = 1; off < 16; off <<= 1)
                rs[r] += __shfl_xor(rs[r], off, 32);
            lrow[r] = lrow[r] * alpha[r] + rs[r];
            mrow[r] = mnew[r];
        }
#pragma unroll
        for (int oj = 0; oj < 8; ++oj)
#pragma unroll
            for (int r = 0; r < 8; ++r) oacc[oj][r] *= alpha[r];

        // write P (bf16) in A-layout rows
#pragma unroll
        for (int sj = 0; sj < 8; ++sj)
#pragma unroll
            for (int r = 0; r < 8; ++r)
                Ps[(qr0 + r + half8) * 128 + sj * 16 + (lane & 15)] =
                    f2bf(sacc[sj][r]);
        __syncthreads();

        // O += P * V
        bf16x16 ap[4];
#pragma unroll
        for (int ks = 0; ks < 4; ++ks)
            ap[ks] = load_a_frag(Ps, qr0, ks * 32, 128, lane);
#pragma unroll
        for (int oj = 0; oj < 8; ++oj)
#pragma unroll
            for (int ks = 0; ks < 4; ++ks) {
                bf16x16 bv = load_b_frag(Vs, oj * 16, ks * 32, 128, lane);
                oacc[oj] = wmma_bf16(ap[ks], bv, oacc[oj]);
            }
    }

    float invl[8];
#pragma unroll
    for (int r = 0; r < 8; ++r) invl[r] = 1.0f / lrow[r];
#pragma unroll
    for (int oj = 0; oj < 8; ++oj)
#pragma unroll
        for (int r = 0; r < 8; ++r) {
            const int sg  = qt * 128 + qr0 + r + half8;
            const int col = h * CHD + oj * 16 + (lane & 15);
            aws[((size_t)(bb * CS + sg)) * (CH * CHD) + col] =
                f2bf(oacc[oj][r] * invl[r]);
        }
}

// ---------------------------------------------------------------------------
// Kernel 4: output projection. A tile (bf16) via TDM, double-buffered LDS so
// the DMA of tile k+1 overlaps WMMA on tile k. B tile register-pipelined.
// ---------------------------------------------------------------------------
__global__ __launch_bounds__(256)
void qo_gemm(const unsigned short* __restrict__ Aattn,
             const float* __restrict__ Wo,
             float* __restrict__ out) {
    __shared__ unsigned short As[2][128 * 32];
    __shared__ unsigned short Bs[128 * 32];

    const int t    = threadIdx.x;
    const int lane = t & 31;
    const int wid  = t >> 5;
    const int wr   = wid >> 1;
    const int wc   = wid & 1;
    const int m0   = blockIdx.y * 128;
    const int n0   = blockIdx.x * 128;

    float4 fb[4];
    auto loadB = [&](int kk) {
#pragma unroll
        for (int i = 0; i < 4; ++i) {
            const int idx = t + i * 256;
            const int row = idx >> 5, c4 = idx & 31;
            fb[i] = *(const float4*)(Wo + (size_t)(kk + row) * CD + n0 + c4 * 4);
        }
    };
    auto storeB = [&]() {
#pragma unroll
        for (int i = 0; i < 4; ++i) {
            const int idx = t + i * 256;
            const int row = idx >> 5, c4 = idx & 31;
            const int cb = c4 * 4;
            Bs[(cb + 0) * 32 + row] = f2bf(fb[i].x);
            Bs[(cb + 1) * 32 + row] = f2bf(fb[i].y);
            Bs[(cb + 2) * 32 + row] = f2bf(fb[i].z);
            Bs[(cb + 3) * 32 + row] = f2bf(fb[i].w);
        }
    };
#if !HAVE_TDM
    uint4 areg[2];
    auto loadA = [&](int kk) {
#pragma unroll
        for (int i = 0; i < 2; ++i) {
            const int idx = t + i * 256;
            const int row = idx >> 2, cc = idx & 3;
            areg[i] = ((const uint4*)(Aattn + (size_t)(m0 + row) * CQN + kk))[cc];
        }
    };
    auto storeA = [&](int buf) {
#pragma unroll
        for (int i = 0; i < 2; ++i) {
            const int idx = t + i * 256;
            const int row = idx >> 2, cc = idx & 3;
            ((uint4*)(As[buf] + row * 32))[cc] = areg[i];
        }
    };
#endif

    f32x8 c[2][4];
#pragma unroll
    for (int i = 0; i < 2; ++i)
#pragma unroll
        for (int j = 0; j < 4; ++j) c[i][j] = zero8();

#if HAVE_TDM
    if (wid == 0)
        tdm_load_2d(lds_off(&As[0][0]), Aattn + (size_t)m0 * CQN,
                    32, 128, CQN, 128, CQN);
#else
    loadA(0);
#endif
    loadB(0);

    for (int k0 = 0; k0 < CQN; k0 += 32) {
        const int cur = (k0 >> 5) & 1;
#if HAVE_TDM
        if (wid == 0) __builtin_amdgcn_s_wait_tensorcnt(0);
        storeB();
#else
        storeA(cur);
        storeB();
#endif
        __syncthreads();
        if (k0 + 32 < CQN) {
#if HAVE_TDM
            if (wid == 0)
                tdm_load_2d(lds_off(&As[cur ^ 1][0]),
                            Aattn + (size_t)m0 * CQN + k0 + 32,
                            32, 128, CQN, 128, CQN);
#else
            loadA(k0 + 32);
#endif
            loadB(k0 + 32);
        }

        bf16x16 a[2], b[4];
#pragma unroll
        for (int si = 0; si < 2; ++si)
            a[si] = load_a_frag(As[cur], wr * 32 + si * 16, 0, 32, lane);
#pragma unroll
        for (int sj = 0; sj < 4; ++sj)
            b[sj] = load_b_frag(Bs, wc * 64 + sj * 16, 0, 32, lane);
#pragma unroll
        for (int si = 0; si < 2; ++si)
#pragma unroll
            for (int sj = 0; sj < 4; ++sj)
                c[si][sj] = wmma_bf16(a[si], b[sj], c[si][sj]);
        __syncthreads();
    }

    const int half8 = ((lane >> 4) & 1) * 8;
#pragma unroll
    for (int si = 0; si < 2; ++si)
#pragma unroll
        for (int sj = 0; sj < 4; ++sj)
#pragma unroll
            for (int r = 0; r < 8; ++r) {
                const int gm = m0 + wr * 32 + si * 16 + r + half8;
                const int gn = n0 + wc * 64 + sj * 16 + (lane & 15);
                out[(size_t)gm * CD + gn] = c[si][sj][r];
            }
}

// ---------------------------------------------------------------------------
extern "C" void kernel_launch(void* const* d_in, const int* in_sizes, int n_in,
                              void* d_out, int out_size, void* d_ws, size_t ws_size,
                              hipStream_t stream) {
    (void)in_sizes; (void)n_in; (void)out_size; (void)ws_size;
    const float* X    = (const float*)d_in[0];
    const float* sins = (const float*)d_in[1];
    const float* coss = (const float*)d_in[2];
    /* d_in[3] = mask: causal mask computed analytically in-kernel */
    const float* Wq   = (const float*)d_in[4];
    const float* Wk   = (const float*)d_in[5];
    const float* Wv   = (const float*)d_in[6];
    const float* Wo   = (const float*)d_in[7];
    float* out = (float*)d_out;

    unsigned short* ws  = (unsigned short*)d_ws;
    unsigned short* qws = ws;                                 // 16,777,216
    unsigned short* kws = qws + (size_t)CB * CH * CS * CHD;   //  4,194,304
    unsigned short* vws = kws + (size_t)CB * CKVH * CS * CHD; //  4,194,304
    unsigned short* aws = vws + (size_t)CB * CKVH * CS * CHD; // 16,777,216

    qkv_gemm<<<dim3((CQN + 2 * CKVN) / 128, CM / 128), 256, 0, stream>>>(
        X, Wq, Wk, Wv, qws, kws, vws);

    const int rope_threads = (CB * CH * CS * (CHD / 2)) + (CB * CKVH * CS * (CHD / 2));
    rope_kernel<<<rope_threads / 256, 256, 0, stream>>>(qws, kws, sins, coss);

    attn_kernel<<<dim3(CS / 128, CH, CB), 256, 4 * 128 * 128 * sizeof(unsigned short),
                  stream>>>(qws, kws, vws, aws);

    qo_gemm<<<dim3(CD / 128, CM / 128), 256, 0, stream>>>(aws, Wo, out);
}
